// SutraV04_49924699849336
// MI455X (gfx1250) — matrix operand order repaired and checked by reading, not compiled, not run on hardware
//
#include <hip/hip_runtime.h>
#include <math.h>

typedef __bf16 bf16_t;
typedef __attribute__((ext_vector_type(16))) __bf16 v16bf;
typedef __attribute__((ext_vector_type(8)))  __bf16 v8bf;
typedef __attribute__((ext_vector_type(8)))  float  v8f;

__device__ __forceinline__ bf16_t f2bf(float f) {
  unsigned u = __builtin_bit_cast(unsigned, f);
  u += 0x7fffu + ((u >> 16) & 1u);                 // round to nearest even
  unsigned short h = (unsigned short)(u >> 16);
  return __builtin_bit_cast(bf16_t, h);
}

__device__ __forceinline__ float sigmoidf_(float x) { return 1.0f / (1.0f + expf(-x)); }

// ---------------------------------------------------------------------------
// WMMA GEMM on pre-converted bf16 operands.
//   A  : bf16, row-major M x K (lda)
//   Bt : bf16, row-major N x K (ldbt)  == B transposed
//   C  : f32, row-major M x N (ldc), C = A*B (+bias[N]) (+addsrc[M,N])
// M%64==0, N%16==0, K%32==0.
// Block = 256 threads = 8 waves arranged 2x4; each wave owns a 32x32 tile
// (4 accumulators, A/B fragments each reused twice).
// Fragment loads are contiguous 16B runs -> pure global_load_b128.
// ---------------------------------------------------------------------------
__global__ __launch_bounds__(256)
void gemm_wmma(const bf16_t* __restrict__ A, int lda,
               const bf16_t* __restrict__ Bt, int ldbt,
               const float* __restrict__ bias,
               const float* __restrict__ addsrc,
               float* __restrict__ C, int ldc,
               int M, int N, int K)
{
  const int lane = threadIdx.x & 31;
  const int wave = threadIdx.x >> 5;
  const int wm = wave >> 2, wn = wave & 3;
  const int m0 = blockIdx.y * 64 + wm * 32;
  const int n0 = blockIdx.x * 128 + wn * 32;
  if (m0 >= M || n0 >= N) return;                  // wave-uniform guard
  const bool hasN2 = (n0 + 16) < N;

  const int l15 = lane & 15;
  const int grp = lane >> 4;

  // per-lane base pointers (ISA fragment layouts, contiguous K runs)
  const bf16_t* pA0 = A + (size_t)(m0 + l15) * lda + 8 * grp;
  const bf16_t* pA1 = A + (size_t)(m0 + 16 + l15) * lda + 8 * grp;
  const bf16_t* pB0 = Bt + (size_t)(n0 + l15) * ldbt + 8 * grp;
  const bf16_t* pB1 = hasN2 ? Bt + (size_t)(n0 + 16 + l15) * ldbt + 8 * grp : pB0;

  v8f acc00 = {}, acc01 = {}, acc10 = {}, acc11 = {};

  for (int k0 = 0; k0 < K; k0 += 32) {
    union { v16bf v; v8bf h[2]; } a0, a1, b0, b1;
    a0.h[0] = *(const v8bf*)(pA0 + k0);            // K = k0+8g   .. +8
    a0.h[1] = *(const v8bf*)(pA0 + k0 + 16);       // K = k0+16+8g.. +8
    a1.h[0] = *(const v8bf*)(pA1 + k0);
    a1.h[1] = *(const v8bf*)(pA1 + k0 + 16);
    b0.h[0] = *(const v8bf*)(pB0 + k0);            // K = k0+8g .. +8, k0+16+8g .. +8
    b0.h[1] = *(const v8bf*)(pB0 + k0 + 16);
    b1.h[0] = *(const v8bf*)(pB1 + k0);
    b1.h[1] = *(const v8bf*)(pB1 + k0 + 16);
    acc00 = __builtin_amdgcn_wmma_f32_16x16x32_bf16(false, a0.v, false, b0.v,
                                                    (short)0, acc00, false, false);
    acc01 = __builtin_amdgcn_wmma_f32_16x16x32_bf16(false, a0.v, false, b1.v,
                                                    (short)0, acc01, false, false);
    acc10 = __builtin_amdgcn_wmma_f32_16x16x32_bf16(false, a1.v, false, b0.v,
                                                    (short)0, acc10, false, false);
    acc11 = __builtin_amdgcn_wmma_f32_16x16x32_bf16(false, a1.v, false, b1.v,
                                                    (short)0, acc11, false, false);
  }

  auto store = [&](v8f acc, int mb, int nb) {
    int nc = nb + l15;
    float bv = bias ? bias[nc] : 0.0f;
#pragma unroll
    for (int v = 0; v < 8; ++v) {
      int mm = mb + v + 8 * grp;                   // C/D layout: M = v + 8*grp
      float r = acc[v] + bv;
      if (addsrc) r += addsrc[(size_t)mm * ldc + nc];
      C[(size_t)mm * ldc + nc] = r;
    }
  };
  store(acc00, m0, n0);
  if (hasN2) store(acc01, m0, n0 + 16);
  store(acc10, m0 + 16, n0);
  if (hasN2) store(acc11, m0 + 16, n0 + 16);
}

// ---------------------------------------------------------------------------
// Precision conversion kernels (outside the GEMM hot loop)
// ---------------------------------------------------------------------------
__global__ void cvt_bf16(const float* __restrict__ in, bf16_t* __restrict__ out, int n) {
  int i = blockIdx.x * blockDim.x + threadIdx.x;
  if (i < n) out[i] = f2bf(in[i]);
}

// in: K x N row-major f32  ->  out: N x K row-major bf16 (transposed)
__global__ void cvt_tr_bf16(const float* __restrict__ in, bf16_t* __restrict__ out,
                            int K, int N) {
  int i = blockIdx.x * blockDim.x + threadIdx.x;   // over N*K (out index)
  if (i >= N * K) return;
  int kk = i % K; int n = i / K;
  out[i] = f2bf(in[(size_t)kk * N + n]);
}

// ---------------------------------------------------------------------------
// Small helper kernels
// ---------------------------------------------------------------------------
__global__ void fill_kernel(float* p, float val, int n) {
  int i = blockIdx.x * blockDim.x + threadIdx.x;
  if (i < n) p[i] = val;
}

__global__ void embed_kernel(const int* __restrict__ x, const float* __restrict__ emb,
                             const float* __restrict__ pos, float* __restrict__ hin) {
  int i = blockIdx.x * blockDim.x + threadIdx.x;   // over 16384*512
  int d = i & 511; int tok = i >> 9;
  int t = tok & 3;
  int id = x[tok];
  hin[i] = emb[(size_t)id * 512 + d] + pos[t * 512 + d];
}

__global__ void gru_pw(const float* __restrict__ gi, const float* __restrict__ gh,
                       float* __restrict__ h, float* __restrict__ out, int t) {
  int i = blockIdx.x * blockDim.x + threadIdx.x;   // over 4096*512
  int d = i & 511; int row = i >> 9;
  const float* gir = gi + (size_t)(row * 4 + t) * 1536;
  const float* ghr = gh + (size_t)row * 1536;
  float hp = h[i];
  float r = sigmoidf_(gir[d] + ghr[d]);
  float z = sigmoidf_(gir[512 + d] + ghr[512 + d]);
  float nn = tanhf(gir[1024 + d] + r * ghr[1024 + d]);
  float hn = (1.0f - z) * nn + z * hp;
  h[i] = hn;
  out[(size_t)(row * 4 + t) * 512 + d] = hn;
}

// wave-per-row LayerNorm over last dim 512, optional residual, in-place ok
__global__ __launch_bounds__(256)
void ln_kernel(const float* __restrict__ X, const float* __restrict__ resid,
               const float* __restrict__ g, const float* __restrict__ b,
               float* __restrict__ Y, int rows) {
  int row = blockIdx.x * 8 + (threadIdx.x >> 5);
  if (row >= rows) return;
  int lane = threadIdx.x & 31;
  const float* x = X + (size_t)row * 512;
  const float* r = resid ? resid + (size_t)row * 512 : nullptr;
  float s = 0.f, ss = 0.f;
  for (int d = lane; d < 512; d += 32) {
    float v = x[d] + (r ? r[d] : 0.f);
    s += v; ss += v * v;
  }
  for (int o = 16; o; o >>= 1) { s += __shfl_xor(s, o, 32); ss += __shfl_xor(ss, o, 32); }
  float mu = s * (1.0f / 512.0f);
  float var = ss * (1.0f / 512.0f) - mu * mu;
  float inv = rsqrtf(var + 1e-5f);
  float* y = Y + (size_t)row * 512;
  for (int d = lane; d < 512; d += 32) {
    float v = x[d] + (r ? r[d] : 0.f);
    y[d] = (v - mu) * inv * g[d] + b[d];
  }
}

// gates[row,k] = softmax_k( h[row,:] @ kgate_W[:,k] + kgate_b[k] )
__global__ __launch_bounds__(256)
void gates_kernel(const float* __restrict__ h, const float* __restrict__ kgW,
                  const float* __restrict__ kgb, float* __restrict__ gts, int rows) {
  int row = blockIdx.x * 8 + (threadIdx.x >> 5);
  if (row >= rows) return;
  int lane = threadIdx.x & 31;
  const float* hr = h + (size_t)row * 512;
  float a0 = 0, a1 = 0, a2 = 0, a3 = 0;
  for (int d = lane; d < 512; d += 32) {
    float hv = hr[d];
    a0 += hv * kgW[d * 4 + 0]; a1 += hv * kgW[d * 4 + 1];
    a2 += hv * kgW[d * 4 + 2]; a3 += hv * kgW[d * 4 + 3];
  }
  for (int o = 16; o; o >>= 1) {
    a0 += __shfl_xor(a0, o, 32); a1 += __shfl_xor(a1, o, 32);
    a2 += __shfl_xor(a2, o, 32); a3 += __shfl_xor(a3, o, 32);
  }
  a0 += kgb[0]; a1 += kgb[1]; a2 += kgb[2]; a3 += kgb[3];
  float m = fmaxf(fmaxf(a0, a1), fmaxf(a2, a3));
  float e0 = expf(a0 - m), e1 = expf(a1 - m), e2 = expf(a2 - m), e3 = expf(a3 - m);
  float inv = 1.0f / (e0 + e1 + e2 + e3);
  if (lane == 0) {
    gts[row * 4 + 0] = e0 * inv; gts[row * 4 + 1] = e1 * inv;
    gts[row * 4 + 2] = e2 * inv; gts[row * 4 + 3] = e3 * inv;
  }
}

// msgs + pack u = [h | msgs].  S already contains kan_b; T holds neighbor part.
__global__ void msgs_kernel(const float* __restrict__ h, const float* __restrict__ S,
                            const float* __restrict__ Tb, const float* __restrict__ gts,
                            float* __restrict__ u) {
  int i = blockIdx.x * blockDim.x + threadIdx.x;   // over 4096*512
  int d = i & 511; int row = i >> 9;
  int n = row & 1023;                              // index within batch (N=1024)
  float g0 = gts[row * 4 + 0], g1 = gts[row * 4 + 1];
  float g2 = gts[row * 4 + 2], g3 = gts[row * 4 + 3];
  float s0 = S[((size_t)row * 4 + 0) * 512 + d];
  float s1 = S[((size_t)row * 4 + 1) * 512 + d];
  float s2 = S[((size_t)row * 4 + 2) * 512 + d];
  float s3 = S[((size_t)row * 4 + 3) * 512 + d];
  float acc = 0.f;
#pragma unroll
  for (int w = 0; w < 5; ++w) {
    int m = n - (4 - w);
    float t0 = 0, t1 = 0, t2 = 0, t3 = 0;
    if (m >= 0) {
      size_t base = ((size_t)(row - (4 - w)) * 4) * 512 + d;
      t0 = Tb[base]; t1 = Tb[base + 512]; t2 = Tb[base + 1024]; t3 = Tb[base + 1536];
    }
    float p0 = s0 + t0, p1 = s1 + t1, p2 = s2 + t2, p3 = s3 + t3;
    acc += g0 * (p0 * sigmoidf_(p0)) + g1 * (p1 * sigmoidf_(p1)) +
           g2 * (p2 * sigmoidf_(p2)) + g3 * (p3 * sigmoidf_(p3));
  }
  acc *= 0.2f;                                     // mean over 5 windows
  u[(size_t)row * 1024 + d] = h[i];
  u[(size_t)row * 1024 + 512 + d] = acc;
}

__global__ void gelu_kernel(float* p, int n) {
  int i = blockIdx.x * blockDim.x + threadIdx.x;
  if (i < n) { float a = p[i]; p[i] = 0.5f * a * (1.0f + erff(a * 0.70710678118654752f)); }
}

// per-round halting: wave per row, mode 0:halt=0, 1:sigmoid, 2:halt=1
__global__ __launch_bounds__(256)
void halt_kernel(const float* __restrict__ h, const float* __restrict__ hW,
                 const float* __restrict__ hb, float* __restrict__ outp,
                 float* __restrict__ rem, float* __restrict__ hsum,
                 int mode, int step, int rows) {
  int row = blockIdx.x * 8 + (threadIdx.x >> 5);
  if (row >= rows) return;
  int lane = threadIdx.x & 31;
  const float* hr = h + (size_t)row * 512;
  float halt;
  if (mode == 0) halt = 0.f;
  else if (mode == 2) halt = 1.f;
  else {
    float a = 0.f;
    for (int d = lane; d < 512; d += 32) a += hr[d] * hW[d];
    for (int o = 16; o; o >>= 1) a += __shfl_xor(a, o, 32);
    halt = sigmoidf_(a + hb[0]);
  }
  float r0 = rem[row];
  float c = r0 * halt;
  float* op = outp + (size_t)row * 512;
  for (int d = lane; d < 512; d += 32) op[d] += c * hr[d];
  if (lane == 0) {
    rem[row] = r0 * (1.0f - halt);
    atomicAdd(&hsum[step], halt);
  }
}

// top-8 causal selection + softmax over selected scores; wave per query row
__global__ __launch_bounds__(256)
void topk_kernel(const float* __restrict__ sc, float* __restrict__ tw,
                 int* __restrict__ ti, int rows) {
  int row = blockIdx.x * 8 + (threadIdx.x >> 5);  // row = b*1024 + n
  if (row >= rows) return;
  int lane = threadIdx.x & 31;
  int n = row & 1023;
  const float* srow = sc + (size_t)row * 1024;
  const float scale = 0.04419417382415922f;       // 1/sqrt(512)
  float av[8]; int ai[8];
#pragma unroll
  for (int j = 0; j < 8; ++j) { av[j] = -3e38f; ai[j] = -1; }
  for (int m = lane; m <= n; m += 32) {           // causal: m <= n only
    float v = srow[m] * scale;
    if (v > av[7]) {
      int j = 7;
      while (j > 0 && v > av[j - 1]) { av[j] = av[j - 1]; ai[j] = ai[j - 1]; --j; }
      av[j] = v; ai[j] = m;
    }
  }
  // merge per-lane sorted lists: 8 rounds of wave arg-max
  int ptr = 0;
  float topv[8]; int topm[8];
#pragma unroll
  for (int i = 0; i < 8; ++i) {
    float hv = (ptr < 8) ? av[ptr] : -3e38f;
    int   hm = (ptr < 8) ? ai[ptr] : -1;
    float bv = hv; int bm = hm;
    for (int o = 16; o; o >>= 1) {
      float ov = __shfl_xor(bv, o, 32);
      int   om = __shfl_xor(bm, o, 32);
      if (ov > bv || (ov == bv && om > bm)) { bv = ov; bm = om; }
    }
    if (ptr < 8 && hm == bm && hm >= 0) ++ptr;
    topv[i] = bv; topm[i] = bm;
  }
  float mx = topv[0];
  float e[8]; float s = 0.f;
#pragma unroll
  for (int i = 0; i < 8; ++i) { e[i] = (topm[i] >= 0) ? expf(topv[i] - mx) : 0.f; s += e[i]; }
  float inv = 1.0f / fmaxf(s, 1e-30f);
  if (lane == 0) {
#pragma unroll
    for (int i = 0; i < 8; ++i) { tw[row * 8 + i] = e[i] * inv; ti[row * 8 + i] = topm[i]; }
  }
}

// retrieved[row,:] = sum_j w[j] * v[b*1024 + idx[j], :]
__global__ __launch_bounds__(256)
void retrieve_kernel(const float* __restrict__ vbuf, const float* __restrict__ tw,
                     const int* __restrict__ ti, float* __restrict__ ret, int rows) {
  int row = blockIdx.x * 8 + (threadIdx.x >> 5);
  if (row >= rows) return;
  int lane = threadIdx.x & 31;
  int b = row >> 10;
  float w[8]; int ix[8];
#pragma unroll
  for (int j = 0; j < 8; ++j) { w[j] = tw[row * 8 + j]; ix[j] = ti[row * 8 + j]; }
  for (int d = lane; d < 512; d += 32) {
    float acc = 0.f;
#pragma unroll
    for (int j = 0; j < 8; ++j)
      if (ix[j] >= 0) acc += w[j] * vbuf[((size_t)(b * 1024 + ix[j])) * 512 + d];
    ret[(size_t)row * 512 + d] = acc;
  }
}

// fin = local + sigmoid(gate_pre) * broad_shifted   (in-place on gate_pre)
__global__ void final_pw(const float* __restrict__ local, const float* __restrict__ broad,
                         float* __restrict__ gpre) {
  int i = blockIdx.x * blockDim.x + threadIdx.x;   // over 16384*512
  int d = i & 511; int tkn = i >> 9;
  int rowBN = tkn >> 2;
  int n = rowBN & 1023;
  float bs = (n > 0) ? broad[(size_t)(rowBN - 1) * 512 + d] : 0.f;
  float gt = sigmoidf_(gpre[i]);
  gpre[i] = local[i] + gt * bs;
}

__global__ void kl_kernel(const float* __restrict__ hsum, float* __restrict__ out) {
  if (threadIdx.x != 0 || blockIdx.x != 0) return;
  float kl = 0.f;
#pragma unroll
  for (int s = 0; s < 4; ++s) {
    float p = 0.2f * powf(0.8f, (float)s);
    float mean = hsum[s] * (1.0f / 4096.0f);
    kl += p * (logf(p) - logf(mean + 1e-8f));
  }
  out[0] = kl * 0.25f;
}

// ---------------------------------------------------------------------------
extern "C" void kernel_launch(void* const* d_in, const int* in_sizes, int n_in,
                              void* d_out, int out_size, void* d_ws, size_t ws_size,
                              hipStream_t stream) {
  (void)in_sizes; (void)n_in; (void)out_size; (void)ws_size;
  const int Bb = 4, Np = 1024, D = 512, BN = 4096, TOK = 16384, V = 256;

  const int*   x      = (const int*)  d_in[0];
  const float* emb    = (const float*)d_in[1];
  const float* pos    = (const float*)d_in[2];
  const float* Wih    = (const float*)d_in[3];
  const float* Whh    = (const float*)d_in[4];
  const float* bih    = (const float*)d_in[5];
  const float* bhh    = (const float*)d_in[6];
  const float* ppg    = (const float*)d_in[7];
  const float* ppb    = (const float*)d_in[8];
  const float* sumW   = (const float*)d_in[9];
  const float* sumB   = (const float*)d_in[10];
  const float* kanW   = (const float*)d_in[11];
  const float* kanB   = (const float*)d_in[12];
  const float* kgW    = (const float*)d_in[13];
  const float* kgB    = (const float*)d_in[14];
  const float* uW1    = (const float*)d_in[15];
  const float* uB1    = (const float*)d_in[16];
  const float* uW2    = (const float*)d_in[17];
  const float* uB2    = (const float*)d_in[18];
  const float* haltW  = (const float*)d_in[19];
  const float* haltB  = (const float*)d_in[20];
  const float* mpg    = (const float*)d_in[21];
  const float* mpb    = (const float*)d_in[22];
  const float* qW     = (const float*)d_in[23];
  const float* qB     = (const float*)d_in[24];
  const float* kW     = (const float*)d_in[25];
  const float* kB     = (const float*)d_in[26];
  const float* vW     = (const float*)d_in[27];
  const float* vB     = (const float*)d_in[28];
  const float* oW     = (const float*)d_in[29];
  const float* oB     = (const float*)d_in[30];
  const float* bcW    = (const float*)d_in[31];
  const float* bcB    = (const float*)d_in[32];
  const float* gpW    = (const float*)d_in[33];
  const float* gpB    = (const float*)d_in[34];
  const float* flg    = (const float*)d_in[35];
  const float* flb    = (const float*)d_in[36];
  const float* headW  = (const float*)d_in[37];

  // ---- f32 workspace ----
  float* wsf = (float*)d_ws;
  size_t off = 0;
  auto alloc = [&](size_t n) { float* p = wsf + off; off += n; return p; };
  float* h_in = alloc((size_t)TOK * D);        // h_in, then GRU outputs -> 'local'
  float* gi   = alloc((size_t)TOK * 3 * D);    // (BN,P,3D); reused for scores + finbuf
  float* gh   = alloc((size_t)BN * 3 * D);     // reused later for q,k,v
  float* h    = alloc((size_t)BN * D);         // GRU hidden -> hT (states)
  float* summ = alloc((size_t)BN * D);         // summaries; message-passing h (in place)
  float* S    = alloc((size_t)BN * 4 * D);
  float* Tb   = alloc((size_t)BN * 4 * D);
  float* gts  = alloc((size_t)BN * 4);
  float* u    = alloc((size_t)BN * 2 * D);     // reused later: ret | comb
  float* a1   = alloc((size_t)BN * D);         // reused later: broad
  float* upd  = alloc((size_t)BN * D);         // reused later: topw/topi
  float* outp = alloc((size_t)BN * D);
  float* rem  = alloc((size_t)BN);
  float* hsum = alloc(8);
  // f32 aliases (dead-buffer reuse)
  float* qb    = gh;
  float* kb    = gh + (size_t)BN * D;
  float* vb    = gh + (size_t)2 * BN * D;
  float* sc    = gi;                           // 4*1024*1024 floats
  float* gpre  = gi + (size_t)5 * 1024 * 1024; // 16384*512 floats, fits in gi
  float* tw    = upd;
  int*   ti    = (int*)(upd + (size_t)BN * 8);
  float* ret   = u;
  float* comb  = u + (size_t)BN * D;
  float* broad = a1;

  // ---- bf16 workspace (after f32 pool; 16B-aligned by construction) ----
  bf16_t* bbase = (bf16_t*)(wsf + off);
  size_t offb = 0;
  auto balloc = [&](size_t n) { bf16_t* p = bbase + offb; offb += n; return p; };
  bf16_t* WihT  = balloc((size_t)1536 * 512);
  bf16_t* WhhT  = balloc((size_t)1536 * 512);
  bf16_t* sumWT = balloc((size_t)512 * 512);
  bf16_t* kanWT = balloc((size_t)4 * 1024 * 512);   // [k][top|bot] each 512x512
  bf16_t* uW1T  = balloc((size_t)512 * 1024);
  bf16_t* uW2T  = balloc((size_t)512 * 512);
  bf16_t* qWT   = balloc((size_t)512 * 512);
  bf16_t* kWT   = balloc((size_t)512 * 512);
  bf16_t* vWT   = balloc((size_t)512 * 512);
  bf16_t* oWT   = balloc((size_t)512 * 512);
  bf16_t* bcWT  = balloc((size_t)512 * 512);
  bf16_t* gpWT  = balloc((size_t)512 * 512);
  bf16_t* headWT= balloc((size_t)256 * 512);
  bf16_t* bfA   = balloc((size_t)TOK * D);          // A-operand scratch (max 16384x512)
  bf16_t* bfB   = balloc((size_t)BN * D);           // second operand (k for scores)

  float* logits = (float*)d_out;
  float* klout  = logits + (size_t)TOK * V;

  dim3 blk(256);
  auto gemm = [&](const bf16_t* A, int lda, const bf16_t* Bt, int ldbt,
                  const float* bias, const float* addsrc, float* C, int ldc,
                  int M, int N, int K) {
    dim3 grid((N + 127) / 128, (M + 63) / 64);
    gemm_wmma<<<grid, blk, 0, stream>>>(A, lda, Bt, ldbt, bias, addsrc, C, ldc, M, N, K);
  };
  auto cvt = [&](const float* src, bf16_t* dst, int n) {
    cvt_bf16<<<(n + 255) / 256, blk, 0, stream>>>(src, dst, n);
  };
  auto cvtT = [&](const float* src, bf16_t* dst, int K, int N) {
    cvt_tr_bf16<<<(N * K + 255) / 256, blk, 0, stream>>>(src, dst, K, N);
  };

  // ---- Stage 0: weights -> bf16, transposed to N x K ----
  cvtT(Wih, WihT, 512, 1536);
  cvtT(Whh, WhhT, 512, 1536);
  cvtT(sumW, sumWT, 512, 512);
  for (int k4 = 0; k4 < 4; ++k4) {
    cvtT(kanW + (size_t)k4 * 524288,          kanWT + (size_t)k4 * 524288,          512, 512);
    cvtT(kanW + (size_t)k4 * 524288 + 262144, kanWT + (size_t)k4 * 524288 + 262144, 512, 512);
  }
  cvtT(uW1, uW1T, 1024, 512);
  cvtT(uW2, uW2T, 512, 512);
  cvtT(qW, qWT, 512, 512);  cvtT(kW, kWT, 512, 512);  cvtT(vW, vWT, 512, 512);
  cvtT(oW, oWT, 512, 512);  cvtT(bcW, bcWT, 512, 512); cvtT(gpW, gpWT, 512, 512);
  cvtT(headW, headWT, 512, 256);

  // ---- Stage 1: embedding + GRU ----
  embed_kernel<<<(TOK * D) / 256, blk, 0, stream>>>(x, emb, pos, h_in);
  cvt(h_in, bfA, TOK * D);
  gemm(bfA, D, WihT, D, bih, nullptr, gi, 3 * D, TOK, 3 * D, D);
  fill_kernel<<<(BN * D) / 256, blk, 0, stream>>>(h, 0.f, BN * D);
  for (int t = 0; t < 4; ++t) {
    cvt(h, bfA, BN * D);
    gemm(bfA, D, WhhT, D, bhh, nullptr, gh, 3 * D, BN, 3 * D, D);
    gru_pw<<<(BN * D) / 256, blk, 0, stream>>>(gi, gh, h, h_in, t);
  }
  ln_kernel<<<TOK / 8, blk, 0, stream>>>(h_in, nullptr, ppg, ppb, h_in, TOK); // -> local
  cvt(h, bfA, BN * D);
  gemm(bfA, D, sumWT, D, sumB, nullptr, summ, D, BN, D, D);                   // summaries

  // ---- Stage 2: adaptive-halt message passing ----
  fill_kernel<<<(BN * D) / 256, blk, 0, stream>>>(outp, 0.f, BN * D);
  fill_kernel<<<(BN + 255) / 256, blk, 0, stream>>>(rem, 1.f, BN);
  fill_kernel<<<1, blk, 0, stream>>>(hsum, 0.f, 8);
  for (int s = 0; s < 4; ++s) {
    cvt(summ, bfA, BN * D);
    for (int k4 = 0; k4 < 4; ++k4) {  // S = h @ W_top + kan_b ; T = h @ W_bot
      gemm(bfA, D, kanWT + (size_t)k4 * 524288, D, kanB + k4 * 512,
           nullptr, S + k4 * 512, 4 * D, BN, D, D);
      gemm(bfA, D, kanWT + (size_t)k4 * 524288 + 262144, D,
           nullptr, nullptr, Tb + k4 * 512, 4 * D, BN, D, D);
    }
    gates_kernel<<<BN / 8, blk, 0, stream>>>(summ, kgW, kgB, gts, BN);
    msgs_kernel<<<(BN * D) / 256, blk, 0, stream>>>(summ, S, Tb, gts, u);
    cvt(u, bfA, BN * 2 * D);
    gemm(bfA, 2 * D, uW1T, 2 * D, uB1, nullptr, a1, D, BN, D, 2 * D);
    gelu_kernel<<<(BN * D) / 256, blk, 0, stream>>>(a1, BN * D);
    cvt(a1, bfA, BN * D);
    gemm(bfA, D, uW2T, D, uB2, nullptr, upd, D, BN, D, D);
    ln_kernel<<<BN / 8, blk, 0, stream>>>(summ, upd, mpg, mpb, summ, BN);
    int mode = (s < 2) ? 0 : ((s == 3) ? 2 : 1);
    halt_kernel<<<BN / 8, blk, 0, stream>>>(summ, haltW, haltB, outp, rem, hsum, mode, s, BN);
  }

  // ---- Stage 3: sparse top-k causal retrieval ----
  cvt(outp, bfA, BN * D);
  gemm(bfA, D, qWT, D, qB, nullptr, qb, D, BN, D, D);
  gemm(bfA, D, kWT, D, kB, nullptr, kb, D, BN, D, D);
  gemm(bfA, D, vWT, D, vB, nullptr, vb, D, BN, D, D);
  cvt(qb, bfA, BN * D);    // q as A operand
  cvt(kb, bfB, BN * D);    // k rows are already the N x K (transposed-B) layout
  for (int b = 0; b < Bb; ++b)
    gemm(bfA + (size_t)b * Np * D, D, bfB + (size_t)b * Np * D, D,
         nullptr, nullptr, sc + (size_t)b * Np * Np, Np, Np, Np, D);
  topk_kernel<<<BN / 8, blk, 0, stream>>>(sc, tw, ti, BN);
  retrieve_kernel<<<BN / 8, blk, 0, stream>>>(vb, tw, ti, ret, BN);
  cvt(ret, bfA, BN * D);
  gemm(bfA, D, oWT, D, oB, outp, comb, D, BN, D, D);   // combined = output + ret@oW+oB

  // ---- Stage 4: broadcast / gate / final LN / head ----
  cvt(comb, bfA, BN * D);
  gemm(bfA, D, bcWT, D, bcB, nullptr, broad, D, BN, D, D);
  cvt(h_in, bfA, TOK * D);
  gemm(bfA, D, gpWT, D, gpB, nullptr, gpre, D, TOK, D, D);
  final_pw<<<(TOK * D) / 256, blk, 0, stream>>>(h_in, broad, gpre);
  ln_kernel<<<TOK / 8, blk, 0, stream>>>(gpre, nullptr, flg, flb, gpre, TOK);
  cvt(gpre, bfA, TOK * D);
  gemm(bfA, D, headWT, D, nullptr, nullptr, logits, V, TOK, V, D);

  kl_kernel<<<1, 32, 0, stream>>>(hsum, klout);
}
